// VariationalDimenetEncoder_25580825215188
// MI455X (gfx1250) — matrix operand order repaired and checked by reading, not compiled, not run on hardware
//
#include <hip/hip_runtime.h>
#include <math.h>

typedef __attribute__((ext_vector_type(16))) _Float16 v16h;
typedef __attribute__((ext_vector_type(8)))  float    v8f;

#define PI_D 3.14159265358979323846

__device__ __forceinline__ float siluf(float x){ return x / (1.0f + __expf(-x)); }

__device__ float jl_f(float x, int l){
    float jm = __sinf(x) / x;
    if (l == 0) return jm;
    float jc = jm / x - __cosf(x) / x;
    for (int n = 1; n < l; n++){ float t = jc; jc = (2.0f*n + 1.0f) / x * jc - jm; jm = t; }
    return jc;
}

__device__ double jl_d(double x, int l){
    double jm = sin(x) / x;
    if (l == 0) return jm;
    double jc = jm / x - cos(x) / x;
    for (int n = 1; n < l; n++){ double t = jc; jc = (2.0*n + 1.0) / x * jc - jm; jm = t; }
    return jc;
}

// ---- constants: spherical Bessel zeros (bisection, as in reference), norms, Y coef ----
__global__ void init_consts_kernel(float* Z, float* NRM, float* YC){
    if (threadIdx.x != 0 || blockIdx.x != 0) return;
    double roots[13];
    for (int i = 0; i < 13; i++) roots[i] = PI_D * (double)(i + 1);
    for (int n = 0; n < 6; n++) Z[n] = (float)roots[n];
    for (int l = 1; l < 7; l++){
        int cnt = 13 - l;
        double lo[13], hi[13];
        for (int i = 0; i < cnt; i++){ lo[i] = roots[i]; hi[i] = roots[i + 1]; }
        for (int it = 0; it < 80; it++){
            for (int i = 0; i < cnt; i++){
                double mid = 0.5 * (lo[i] + hi[i]);
                bool same = (jl_d(lo[i], l) > 0.0) == (jl_d(mid, l) > 0.0);
                if (same) lo[i] = mid; else hi[i] = mid;
            }
        }
        for (int i = 0; i < cnt; i++) roots[i] = 0.5 * (lo[i] + hi[i]);
        for (int n = 0; n < 6; n++) Z[l * 6 + n] = (float)roots[n];
    }
    for (int l = 0; l < 7; l++)
        for (int n = 0; n < 6; n++)
            NRM[l * 6 + n] = (float)(sqrt(2.0) / fabs(jl_d((double)Z[l * 6 + n], l + 1)));
    for (int l = 0; l < 7; l++) YC[l] = (float)sqrt((2.0 * l + 1.0) / (4.0 * PI_D));
}

__global__ __launch_bounds__(256) void zero_kernel(float* p, long n){
    long i = (long)blockIdx.x * blockDim.x + threadIdx.x;
    if (i < n) p[i] = 0.0f;
}

// ---- per-edge geometry: dist, envelope, rbf[6], rbf_sph[7][6] ----
__global__ __launch_bounds__(256) void edge_geom_kernel(
    const float* __restrict__ pos, const int* __restrict__ src, const int* __restrict__ dst,
    const float* __restrict__ freq, const float* __restrict__ Z, const float* __restrict__ NRM,
    float* __restrict__ rbf, float* __restrict__ rbfsph, int E)
{
    int e = blockIdx.x * blockDim.x + threadIdx.x;
    if (e >= E) return;
    int i = dst[e], j = src[e];
    float dx = pos[i*3+0]-pos[j*3+0], dy = pos[i*3+1]-pos[j*3+1], dz = pos[i*3+2]-pos[j*3+2];
    float d = sqrtf(dx*dx + dy*dy + dz*dz);
    float xd = d * (1.0f / 5.0f);
    float env;
    if (xd < 1.0f){
        float x2 = xd*xd, x4 = x2*x2, x5 = x4*xd, x6 = x5*xd, x7 = x6*xd;
        env = 1.0f/xd - 28.0f*x5 + 48.0f*x6 - 21.0f*x7;
    } else env = 0.0f;
    for (int r = 0; r < 6; r++) rbf[(size_t)e*6 + r] = env * __sinf(freq[r] * xd);
    for (int l = 0; l < 7; l++)
        for (int r = 0; r < 6; r++){
            float x = xd * Z[l*6 + r];
            rbfsph[(size_t)e*42 + l*6 + r] = NRM[l*6 + r] * jl_f(x, l) * env;
        }
}

// ---- build [E,192] concat: [hz_i | hz_j | silu(rbf@Wr+br)] ----
__global__ __launch_bounds__(256) void build_cat_kernel(
    const float* __restrict__ rbf, const float* __restrict__ Wr, const float* __restrict__ br,
    const float* __restrict__ emb, const int* __restrict__ z,
    const int* __restrict__ src, const int* __restrict__ dst,
    float* __restrict__ cat, int E)
{
    long idx = (long)blockIdx.x * blockDim.x + threadIdx.x;
    if (idx >= (long)E * 64) return;
    int e = (int)(idx >> 6), h = (int)(idx & 63);
    float a = br[h];
    for (int r = 0; r < 6; r++) a += rbf[(size_t)e*6 + r] * Wr[r*64 + h];
    a = siluf(a);
    cat[(size_t)e*192 + 128 + h] = a;
    cat[(size_t)e*192 +   0 + h] = emb[(size_t)z[dst[e]]*64 + h];
    cat[(size_t)e*192 +  64 + h] = emb[(size_t)z[src[e]]*64 + h];
}

// ---- per-triplet: angle -> cbf -> sbf[42] ----
__global__ __launch_bounds__(256) void triplet_kernel(
    const float* __restrict__ pos, const int* __restrict__ esrc, const int* __restrict__ edst,
    const int* __restrict__ tkj, const int* __restrict__ tji,
    const float* __restrict__ rbfsph, const float* __restrict__ YC,
    float* __restrict__ sbf, int T)
{
    int t = blockIdx.x * blockDim.x + threadIdx.x;
    if (t >= T) return;
    int ji = tji[t], kj = tkj[t];
    int ti = edst[ji], tj = esrc[ji], tk = esrc[kj];
    float ax = pos[ti*3+0]-pos[tj*3+0], ay = pos[ti*3+1]-pos[tj*3+1], az = pos[ti*3+2]-pos[tj*3+2];
    float bx = pos[tk*3+0]-pos[tj*3+0], by = pos[tk*3+1]-pos[tj*3+1], bz = pos[tk*3+2]-pos[tj*3+2];
    float dt = ax*bx + ay*by + az*bz;
    float cx = ay*bz - az*by, cy = az*bx - ax*bz, cz = ax*by - ay*bx;
    float cr2 = cx*cx + cy*cy + cz*cz;
    float c = dt * rsqrtf(fmaxf(dt*dt + cr2, 1e-30f));
    float c2 = c*c, c3 = c2*c, c4 = c2*c2, c5 = c4*c, c6 = c4*c2;
    float cbf[7];
    cbf[0] = YC[0];
    cbf[1] = YC[1] * c;
    cbf[2] = YC[2] * 0.5f * (3.0f*c2 - 1.0f);
    cbf[3] = YC[3] * 0.5f * (5.0f*c3 - 3.0f*c);
    cbf[4] = YC[4] * 0.125f * (35.0f*c4 - 30.0f*c2 + 3.0f);
    cbf[5] = YC[5] * 0.125f * (63.0f*c5 - 70.0f*c3 + 15.0f*c);
    cbf[6] = YC[6] * (231.0f*c6 - 315.0f*c4 + 105.0f*c2 - 5.0f) * (1.0f/16.0f);
    for (int l = 0; l < 7; l++)
        for (int r = 0; r < 6; r++)
            sbf[(size_t)t*42 + l*6 + r] = rbfsph[(size_t)kj*42 + l*6 + r] * cbf[l];
}

// ---- sbf @ W_sbf : [T,42]x[42,8] ----
__global__ __launch_bounds__(256) void sbf_proj_kernel(
    const float* __restrict__ sbf, const float* __restrict__ Ws, float* __restrict__ sbfb, int T)
{
    long idx = (long)blockIdx.x * blockDim.x + threadIdx.x;
    if (idx >= (long)T * 8) return;
    int t = (int)(idx >> 3), b = (int)(idx & 7);
    float a = 0.0f;
    for (int k = 0; k < 42; k++) a += sbf[(size_t)t*42 + k] * Ws[k*8 + b];
    sbfb[idx] = a;
}

// ---- x_kj = t1 * (rbf @ W_rbf) ----
__global__ __launch_bounds__(256) void mul_rbf_kernel(
    const float* __restrict__ t1, const float* __restrict__ rbf,
    const float* __restrict__ Wr, float* __restrict__ out, int E)
{
    long idx = (long)blockIdx.x * blockDim.x + threadIdx.x;
    if (idx >= (long)E * 64) return;
    int e = (int)(idx >> 6), h = (int)(idx & 63);
    float a = 0.0f;
    for (int r = 0; r < 6; r++) a += rbf[(size_t)e*6 + r] * Wr[r*64 + h];
    out[idx] = t1[idx] * a;
}

// ---- generic WMMA GEMM: OUT[r,n] (=|+=) [addsrc +] act( X[r,:K] @ W[K,N] + bias ) ----
// one wave (32 lanes) per 16-row tile; KDIM compile-time (64 or 192), N multiple of 16.
// Branch-free epilogue via template flags; full-tile fast path.
template<int KDIM, int ACT, int HAS_ADD, int ACCUM>
__global__ __launch_bounds__(32) void gemm_wmma_kernel(
    const float* __restrict__ X, const float* __restrict__ W,
    const float* __restrict__ bias, const float* __restrict__ addsrc,
    float* __restrict__ OUT, int rows, int ncols)
{
    __shared__ float ldsX[16 * KDIM];
    const int lane = threadIdx.x;
    const int r0 = blockIdx.x * 16;
    const int m  = lane & 15;
    const int hi = lane >> 4;
    constexpr int HALF = KDIM / 2;      // 32 or 96 floats, 16B-aligned chunks

    // stage X tile [16, KDIM] into LDS with b128 moves (zero-padded past `rows`)
    {
        const int lr = lane >> 1;
        const int c0 = (lane & 1) * HALF;
        const int grow = r0 + lr;
        float4* dstv = (float4*)&ldsX[lr * KDIM + c0];
        if (grow < rows){
            const float4* srcv = (const float4*)(X + (size_t)grow * KDIM + c0);
            #pragma unroll
            for (int q = 0; q < HALF / 4; q++) dstv[q] = srcv[q];
        } else {
            float4 zz = make_float4(0.f, 0.f, 0.f, 0.f);
            #pragma unroll
            for (int q = 0; q < HALF / 4; q++) dstv[q] = zz;
        }
    }
    __builtin_prefetch(W + (size_t)lane * ncols, 0, 1);   // global_prefetch_b8 on W panel
    __syncthreads();

    const int koff = hi ? 8 : 0;
    const int nts = ncols >> 4;
    const bool full = (r0 + 16) <= rows;
    for (int nt = 0; nt < nts; nt++){
        const int n = nt * 16 + m;
        v8f acc = {};
        #pragma unroll
        for (int ks = 0; ks < KDIM / 32; ks++){
            const int k0 = ks * 32;
            v16h av, bv;
            #pragma unroll
            for (int e = 0; e < 8; e++){
                av[e]     = (_Float16)ldsX[m * KDIM + k0 + koff + e];
                av[8 + e] = (_Float16)ldsX[m * KDIM + k0 + 16 + koff + e];
            }
            #pragma unroll
            for (int e = 0; e < 8; e++){
                bv[e]     = (_Float16)W[(size_t)(k0 + koff + e) * ncols + n];
                bv[8 + e] = (_Float16)W[(size_t)(k0 + 16 + koff + e) * ncols + n];
            }
            acc = __builtin_amdgcn_wmma_f32_16x16x32_f16(false, av, false, bv,
                                                         (short)0, acc, false, false);
        }
        const float bb = bias ? bias[n] : 0.0f;
        if (full){
            #pragma unroll
            for (int r = 0; r < 8; r++){
                const int rr = r0 + r + hi * 8;
                float v = acc[r] + bb;
                if (ACT) v = siluf(v);
                const size_t oidx = (size_t)rr * ncols + n;
                if (HAS_ADD) v += addsrc[oidx];
                if (ACCUM) OUT[oidx] += v; else OUT[oidx] = v;
            }
        } else {
            #pragma unroll
            for (int r = 0; r < 8; r++){
                const int rr = r0 + r + hi * 8;
                if (rr < rows){
                    float v = acc[r] + bb;
                    if (ACT) v = siluf(v);
                    const size_t oidx = (size_t)rr * ncols + n;
                    if (HAS_ADD) v += addsrc[oidx];
                    if (ACCUM) OUT[oidx] += v; else OUT[oidx] = v;
                }
            }
        }
    }
}

// ---- bilinear triplet message via WMMA:
//   msg[t,o] = sum_{b,h} sbfb[t,b]*xkj[trip_kj[t],h]*Wbil[o, b*64+h]  (K=512)
//   scatter:  hbuf[trip_ji[t], o] += msg[t,o]  (atomic)
__global__ __launch_bounds__(32) void msg_wmma_kernel(
    const float* __restrict__ sbfb, const float* __restrict__ xkj,
    const float* __restrict__ wbil,
    const int* __restrict__ tkj, const int* __restrict__ tji,
    float* __restrict__ hbuf, int T)
{
    __shared__ float ldsX[16 * 64];
    __shared__ float ldsS[16 * 8];
    const int lane = threadIdx.x;
    const int t0 = blockIdx.x * 16;
    const int m  = lane & 15;
    const int hi = lane >> 4;

    // stage gathered x_kj rows (b128 moves)
    {
        const int lr = lane >> 1;
        const int c0 = (lane & 1) * 32;
        const int t2 = t0 + lr;
        float4* dstv = (float4*)&ldsX[lr * 64 + c0];
        if (t2 < T){
            const int e2 = tkj[t2];
            const float4* srcv = (const float4*)(xkj + (size_t)e2 * 64 + c0);
            #pragma unroll
            for (int q = 0; q < 8; q++) dstv[q] = srcv[q];
        } else {
            float4 zz = make_float4(0.f, 0.f, 0.f, 0.f);
            #pragma unroll
            for (int q = 0; q < 8; q++) dstv[q] = zz;
        }
    }
    // stage sbf_b rows
    #pragma unroll
    for (int q = 0; q < 4; q++){
        int idx = lane * 4 + q;
        int rr = idx >> 3, bb = idx & 7;
        int tt = t0 + rr;
        ldsS[idx] = (tt < T) ? sbfb[(size_t)tt * 8 + bb] : 0.0f;
    }
    __builtin_prefetch(wbil + (size_t)lane * 512, 0, 1);
    __syncthreads();

    const int koff = hi ? 8 : 0;
    const bool full = (t0 + 16) <= T;
    for (int nt = 0; nt < 4; nt++){
        const int n = nt * 16 + m;
        v8f acc = {};
        #pragma unroll
        for (int ks = 0; ks < 16; ks++){
            const int k0 = ks * 32;
            const float sv = ldsS[m * 8 + (k0 >> 6)];  // b-chunk constant within a 32-wide k-step
            v16h av, bv;
            #pragma unroll
            for (int e = 0; e < 8; e++){
                const int ka = k0 + koff + e;
                const int kc = k0 + 16 + koff + e;
                av[e]     = (_Float16)(sv * ldsX[m * 64 + (ka & 63)]);
                av[8 + e] = (_Float16)(sv * ldsX[m * 64 + (kc & 63)]);
            }
            #pragma unroll
            for (int e = 0; e < 8; e++){
                bv[e]     = (_Float16)wbil[(size_t)n * 512 + k0 + koff + e];
                bv[8 + e] = (_Float16)wbil[(size_t)n * 512 + k0 + 16 + koff + e];
            }
            acc = __builtin_amdgcn_wmma_f32_16x16x32_f16(false, av, false, bv,
                                                         (short)0, acc, false, false);
        }
        if (full){
            #pragma unroll
            for (int r = 0; r < 8; r++){
                const int tm = t0 + r + hi * 8;
                const int ej = tji[tm];
                atomicAdd(&hbuf[(size_t)ej * 64 + n], acc[r]);
            }
        } else {
            #pragma unroll
            for (int r = 0; r < 8; r++){
                const int tm = t0 + r + hi * 8;
                if (tm < T){
                    const int ej = tji[tm];
                    atomicAdd(&hbuf[(size_t)ej * 64 + n], acc[r]);
                }
            }
        }
    }
}

// ---- out-block gather: hnode[i] += (rbf@Wr) * xe  (scatter by edge_dst) ----
__global__ __launch_bounds__(256) void out_gather_kernel(
    const float* __restrict__ rbf, const float* __restrict__ Wr,
    const float* __restrict__ xe, const int* __restrict__ dst,
    float* __restrict__ hnode, int E)
{
    long idx = (long)blockIdx.x * blockDim.x + threadIdx.x;
    if (idx >= (long)E * 64) return;
    int e = (int)(idx >> 6), h = (int)(idx & 63);
    float a = 0.0f;
    for (int r = 0; r < 6; r++) a += rbf[(size_t)e*6 + r] * Wr[r*64 + h];
    atomicAdd(&hnode[(size_t)dst[e] * 64 + h], a * xe[idx]);
}

// ---- graph pooling ----
__global__ __launch_bounds__(256) void pool_kernel(
    const float* __restrict__ P, const int* __restrict__ batch, float* __restrict__ xg, int N)
{
    long idx = (long)blockIdx.x * blockDim.x + threadIdx.x;
    if (idx >= (long)N * 128) return;
    int n = (int)(idx >> 7), c = (int)(idx & 127);
    atomicAdd(&xg[(size_t)batch[n] * 128 + c], P[idx]);
}

__global__ __launch_bounds__(256) void relu_kernel(float* p, long n){
    long i = (long)blockIdx.x * blockDim.x + threadIdx.x;
    if (i < n) p[i] = fmaxf(p[i], 0.0f);
}

// ---- GAT head (G=49, tiny) ----
__global__ __launch_bounds__(256) void gat_hh_kernel(
    const float* __restrict__ xg, const float* __restrict__ W, float* __restrict__ hh, int G)
{
    long idx = (long)blockIdx.x * blockDim.x + threadIdx.x;
    if (idx >= (long)G * 64) return;
    int g = (int)(idx >> 6), o = (int)(idx & 63);
    float a = 0.0f;
    for (int c = 0; c < 128; c++) a += xg[(size_t)g*128 + c] * W[c*64 + o];
    hh[idx] = a;
}

__global__ __launch_bounds__(64) void gat_att_kernel(
    const float* __restrict__ hh, const float* __restrict__ atts, const float* __restrict__ attd,
    float* __restrict__ as_, float* __restrict__ ad_, int G)
{
    int g = threadIdx.x;
    if (g >= G) return;
    float s = 0.0f, d = 0.0f;
    for (int o = 0; o < 64; o++){ float v = hh[(size_t)g*64 + o]; s += v*atts[o]; d += v*attd[o]; }
    as_[g] = s; ad_[g] = d;
}

__device__ __forceinline__ float leakyf(float x){ return x > 0.0f ? x : 0.2f * x; }

__global__ __launch_bounds__(64) void gat_softmax_kernel(
    const float* __restrict__ as_, const float* __restrict__ ad_,
    const int* __restrict__ gs, const int* __restrict__ gd,
    float* __restrict__ mm, float* __restrict__ dn, int G, int GE)
{
    int g = threadIdx.x;
    if (g >= G) return;
    float m = -1e30f;
    for (int e = 0; e < GE; e++)
        if (gd[e] == g) m = fmaxf(m, leakyf(as_[gs[e]] + ad_[g]));
    float den = 0.0f;
    for (int e = 0; e < GE; e++)
        if (gd[e] == g) den += __expf(leakyf(as_[gs[e]] + ad_[g]) - m);
    mm[g] = m; dn[g] = (den > 0.0f) ? den : 1.0f;
}

__global__ __launch_bounds__(256) void gat_out_kernel(
    const float* __restrict__ hh, const float* __restrict__ as_, const float* __restrict__ ad_,
    const float* __restrict__ mm, const float* __restrict__ dn,
    const int* __restrict__ gs, const int* __restrict__ gd,
    const float* __restrict__ bias, float* __restrict__ outp, int G, int GE)
{
    long idx = (long)blockIdx.x * blockDim.x + threadIdx.x;
    if (idx >= (long)G * 64) return;
    int g = (int)(idx >> 6), o = (int)(idx & 63);
    float acc = 0.0f;
    for (int e = 0; e < GE; e++){
        if (gd[e] == g){
            float v = leakyf(as_[gs[e]] + ad_[g]);
            acc += __expf(v - mm[g]) / dn[g] * hh[(size_t)gs[e] * 64 + o];
        }
    }
    outp[idx] = acc + bias[o];
}

extern "C" void kernel_launch(void* const* d_in, const int* in_sizes, int n_in,
                              void* d_out, int out_size, void* d_ws, size_t ws_size,
                              hipStream_t stream)
{
    (void)n_in; (void)out_size; (void)ws_size;
    const float* pos        = (const float*)d_in[0];
    const float* emb        = (const float*)d_in[1];
    const float* freq       = (const float*)d_in[2];
    const float* emb_W_rbf  = (const float*)d_in[3];
    const float* emb_b_rbf  = (const float*)d_in[4];
    const float* emb_W      = (const float*)d_in[5];
    const float* emb_b      = (const float*)d_in[6];
    const float* int_W_rbf  = (const float*)d_in[7];
    const float* int_W_sbf  = (const float*)d_in[8];
    const float* int_W_bil  = (const float*)d_in[9];
    const float* int_W_kj   = (const float*)d_in[10];
    const float* int_b_kj   = (const float*)d_in[11];
    const float* int_W_ji   = (const float*)d_in[12];
    const float* int_b_ji   = (const float*)d_in[13];
    const float* int_res_W  = (const float*)d_in[14];
    const float* int_res_b  = (const float*)d_in[15];
    const float* int_W_mid  = (const float*)d_in[16];
    const float* int_b_mid  = (const float*)d_in[17];
    const float* out_W_rbf  = (const float*)d_in[18];
    const float* out_W_lin  = (const float*)d_in[19];
    const float* out_b_lin  = (const float*)d_in[20];
    const float* out_W_final= (const float*)d_in[21];
    const float* gat_W      = (const float*)d_in[22];
    const float* gat_att_src= (const float*)d_in[23];
    const float* gat_att_dst= (const float*)d_in[24];
    const float* gat_bias   = (const float*)d_in[25];
    const int* z        = (const int*)d_in[26];
    const int* edge_src = (const int*)d_in[27];
    const int* edge_dst = (const int*)d_in[28];
    const int* trip_kj  = (const int*)d_in[29];
    const int* trip_ji  = (const int*)d_in[30];
    const int* batch    = (const int*)d_in[31];
    const int* g_src    = (const int*)d_in[32];
    const int* g_dst    = (const int*)d_in[33];

    const int N  = in_sizes[26];
    const int E  = in_sizes[27];
    const int T  = in_sizes[29];
    const int GE = in_sizes[32];
    const int G  = 49;

    float* w = (float*)d_ws;
    size_t off = 0;
    auto A = [&](size_t n){ float* p = w + off; off += (n + 63) & ~(size_t)63; return p; };
    float* Zc    = A(42);
    float* NRMc  = A(42);
    float* YCc   = A(16);
    float* rbf    = A((size_t)E * 6);
    float* rbfsph = A((size_t)E * 42);
    float* sbf    = A((size_t)T * 42);
    float* sbfb   = A((size_t)T * 8);
    float* cat    = A((size_t)E * 192);
    float* xe     = A((size_t)E * 64);
    float* bufB   = A((size_t)E * 64);
    float* bufC   = A((size_t)E * 64);
    float* bufD   = A((size_t)E * 64);
    float* hnode  = A((size_t)N * 64);
    float* nodeB  = A((size_t)N * 64);
    float* Pbuf   = A((size_t)N * 128);
    float* xg     = A((size_t)G * 128);
    float* hh     = A((size_t)G * 64);
    float* as_    = A(G);
    float* ad_    = A(G);
    float* mm     = A(G);
    float* dn     = A(G);

    const int TB = 256;
    auto gE  = (unsigned)((E + TB - 1) / TB);
    auto gT  = (unsigned)((T + TB - 1) / TB);
    auto gE64 = (unsigned)(((long)E * 64 + TB - 1) / TB);
    auto gT8  = (unsigned)(((long)T * 8 + TB - 1) / TB);
    auto gN64 = (unsigned)(((long)N * 64 + TB - 1) / TB);
    auto gN128= (unsigned)(((long)N * 128 + TB - 1) / TB);
    auto gG64 = (unsigned)(((long)G * 64 + TB - 1) / TB);
    auto gG128= (unsigned)(((long)G * 128 + TB - 1) / TB);

    // templated dispatch: (kdim, act, has_add, accum) -> one branch-free kernel each
    auto gemm = [&](const float* X, const float* Wp, const float* bp, const float* addp,
                    float* O, int rows, int kdim, int ncols, int act, int accum){
        dim3 g((unsigned)((rows + 15) / 16)), b(32);
        if (kdim == 192){
            gemm_wmma_kernel<192,1,0,0><<<g, b, 0, stream>>>(X, Wp, bp, addp, O, rows, ncols);
        } else if (accum){
            gemm_wmma_kernel<64,0,0,1><<<g, b, 0, stream>>>(X, Wp, bp, addp, O, rows, ncols);
        } else if (addp){
            gemm_wmma_kernel<64,1,1,0><<<g, b, 0, stream>>>(X, Wp, bp, addp, O, rows, ncols);
        } else {
            gemm_wmma_kernel<64,1,0,0><<<g, b, 0, stream>>>(X, Wp, bp, addp, O, rows, ncols);
        }
    };

    auto out_block = [&](int k, const float* xein){
        zero_kernel<<<dim3(gN64), dim3(TB), 0, stream>>>(hnode, (long)N * 64);
        out_gather_kernel<<<dim3(gE64), dim3(TB), 0, stream>>>(
            rbf, out_W_rbf + (size_t)k * 384, xein, edge_dst, hnode, E);
        gemm(hnode, out_W_lin + ((size_t)k*3 + 0) * 4096, out_b_lin + ((size_t)k*3 + 0) * 64,
             nullptr, nodeB, N, 64, 64, 1, 0);
        gemm(nodeB, out_W_lin + ((size_t)k*3 + 1) * 4096, out_b_lin + ((size_t)k*3 + 1) * 64,
             nullptr, hnode, N, 64, 64, 1, 0);
        gemm(hnode, out_W_lin + ((size_t)k*3 + 2) * 4096, out_b_lin + ((size_t)k*3 + 2) * 64,
             nullptr, nodeB, N, 64, 64, 1, 0);
        gemm(nodeB, out_W_final + (size_t)k * 8192, nullptr, nullptr, Pbuf, N, 64, 128, 0, 1);
    };

    // constants + geometry + basis
    init_consts_kernel<<<dim3(1), dim3(1), 0, stream>>>(Zc, NRMc, YCc);
    edge_geom_kernel<<<dim3(gE), dim3(TB), 0, stream>>>(
        pos, edge_src, edge_dst, freq, Zc, NRMc, rbf, rbfsph, E);
    triplet_kernel<<<dim3(gT), dim3(TB), 0, stream>>>(
        pos, edge_src, edge_dst, trip_kj, trip_ji, rbfsph, YCc, sbf, T);

    // embedding: x_e = silu([hz_i|hz_j|rbf_e] @ emb_W + emb_b)
    build_cat_kernel<<<dim3(gE64), dim3(TB), 0, stream>>>(
        rbf, emb_W_rbf, emb_b_rbf, emb, z, edge_src, edge_dst, cat, E);
    gemm(cat, emb_W, emb_b, nullptr, xe, E, 192, 64, 1, 0);

    // P = out_block(0, x_e)
    zero_kernel<<<dim3(gN128), dim3(TB), 0, stream>>>(Pbuf, (long)N * 128);
    out_block(0, xe);

    for (int bk = 0; bk < 4; bk++){
        const float* rW = int_res_W;
        const float* rB = int_res_b;
        auto RW = [&](int r, int io){ return rW + (((size_t)bk*3 + r)*2 + io) * 4096; };
        auto RB = [&](int r, int io){ return rB + (((size_t)bk*3 + r)*2 + io) * 64; };

        sbf_proj_kernel<<<dim3(gT8), dim3(TB), 0, stream>>>(
            sbf, int_W_sbf + (size_t)bk * 336, sbfb, T);

        // bufB = x_ji = silu(x_e@W_ji+b);  bufC = silu(x_e@W_kj+b);  bufD = bufC*(rbf@W_rbf)
        gemm(xe, int_W_ji + (size_t)bk*4096, int_b_ji + (size_t)bk*64, nullptr, bufB, E, 64, 64, 1, 0);
        gemm(xe, int_W_kj + (size_t)bk*4096, int_b_kj + (size_t)bk*64, nullptr, bufC, E, 64, 64, 1, 0);
        mul_rbf_kernel<<<dim3(gE64), dim3(TB), 0, stream>>>(
            bufC, rbf, int_W_rbf + (size_t)bk*384, bufD, E);

        // bufB += scatter(bilinear message)  (WMMA over K=512)
        msg_wmma_kernel<<<dim3((unsigned)((T + 15) / 16)), dim3(32), 0, stream>>>(
            sbfb, bufD, int_W_bil + (size_t)bk*32768, trip_kj, trip_ji, bufB, T);

        // res0: bufD = bufB + silu(silu(bufB@W00+b00)@W01+b01)
        gemm(bufB, RW(0,0), RB(0,0), nullptr, bufC, E, 64, 64, 1, 0);
        gemm(bufC, RW(0,1), RB(0,1), bufB,    bufD, E, 64, 64, 1, 0);
        // mid: bufB = x_e + silu(bufD@W_mid+b_mid)
        gemm(bufD, int_W_mid + (size_t)bk*4096, int_b_mid + (size_t)bk*64, xe, bufB, E, 64, 64, 1, 0);
        // res1: bufD = bufB + silu(silu(bufB@..)@..)
        gemm(bufB, RW(1,0), RB(1,0), nullptr, bufC, E, 64, 64, 1, 0);
        gemm(bufC, RW(1,1), RB(1,1), bufB,    bufD, E, 64, 64, 1, 0);
        // res2 -> new x_e
        gemm(bufD, RW(2,0), RB(2,0), nullptr, bufC, E, 64, 64, 1, 0);
        gemm(bufC, RW(2,1), RB(2,1), bufD,    xe,   E, 64, 64, 1, 0);

        out_block(bk + 1, xe);
    }

    // pooling + relu
    zero_kernel<<<dim3(gG128), dim3(TB), 0, stream>>>(xg, (long)G * 128);
    pool_kernel<<<dim3(gN128), dim3(TB), 0, stream>>>(Pbuf, batch, xg, N);
    relu_kernel<<<dim3(gG128), dim3(TB), 0, stream>>>(xg, (long)G * 128);

    // two GAT heads
    for (int k = 0; k < 2; k++){
        gat_hh_kernel<<<dim3(gG64), dim3(TB), 0, stream>>>(
            xg, gat_W + (size_t)k * 128 * 64, hh, G);
        gat_att_kernel<<<dim3(1), dim3(64), 0, stream>>>(
            hh, gat_att_src + (size_t)k * 64, gat_att_dst + (size_t)k * 64, as_, ad_, G);
        gat_softmax_kernel<<<dim3(1), dim3(64), 0, stream>>>(
            as_, ad_, g_src, g_dst, mm, dn, G, GE);
        gat_out_kernel<<<dim3(gG64), dim3(TB), 0, stream>>>(
            hh, as_, ad_, mm, dn, g_src, g_dst, gat_bias + (size_t)k * 64,
            (float*)d_out + (size_t)k * G * 64, G, GE);
    }
}
